// Conv1dWithLoRA_3358664425876
// MI455X (gfx1250) — matrix operand order repaired and checked by reading, not compiled, not run on hardware
//
#include <hip/hip_runtime.h>

// ---------------------------------------------------------------------------
// Conv1d(256->256, K=5, pad=2) + folded LoRA (rank 8, scaling 2.0), fp32 I/O.
// LoRA folded into the weight: W_eff = W + 2.0 * B @ A  (exact).
// Main conv runs as implicit GEMM on v_wmma_f32_16x16x32_bf16 (f32 accum).
// W_eff is stored pre-swizzled in WMMA A-fragment order so A fetches are
// fully-coalesced global_load_b128 at immediate offsets.
// ---------------------------------------------------------------------------

typedef __attribute__((ext_vector_type(16))) __bf16        v16bf;
typedef __attribute__((ext_vector_type(8)))  float         v8f;
typedef __attribute__((ext_vector_type(4)))  unsigned int  u32x4;

#define C_IN   256
#define C_OUT  256
#define T_LEN  16384
#define KW     5
#define RANK   8
#define KDIM   (C_IN * KW)      // 1280
#define NCHUNK 40               // KDIM / 32
#define NTILE  128              // time columns per block
#define SLAB_T (NTILE + 4)      // 132 (K=5, pad=2 halo)
#define CPAD   264              // padded ci stride in LDS (bank-conflict pad)
#define NBATCH 8
#define SCALING 2.0f            // ALPHA / RANK = 16/8
#define STRIP_STRIDE (NCHUNK * 32 * 16)   // ushorts per 16-row M-strip = 20480

__device__ __forceinline__ unsigned short f32_to_bf16(float f) {
  unsigned int u = __float_as_uint(f);
  u += 0x7FFFu + ((u >> 16) & 1u);          // round to nearest even
  return (unsigned short)(u >> 16);
}

// ---------------------------------------------------------------------------
// Kernel 1: fold LoRA into weight, convert to bf16, store in WMMA A-fragment
// order. Global K ordering is k-major: kidx = k*256 + ci (so each 32-wide
// chunk is 32 consecutive ci at one tap k, matching the LDS im2col slab).
// Fragment layout: W2f[strip=co/16][chunk=kidx/32][lane][elem e=0..15],
//   lane = 16h + (co&15);  K_local = 8h+e (e<8) | 16+8h+(e-8) (e>=8).
// ---------------------------------------------------------------------------
__global__ void fuse_weights_kernel(const float* __restrict__ conv_w,
                                    const float* __restrict__ lora_A,
                                    const float* __restrict__ lora_B,
                                    unsigned short* __restrict__ W2f) {
  const int idx = blockIdx.x * blockDim.x + threadIdx.x;   // [0, 256*1280)
  const int co = idx / KDIM;
  const int r  = idx % KDIM;
  const int k  = r / C_IN;
  const int ci = r % C_IN;

  float acc = conv_w[(co * C_IN + ci) * KW + k];
  float lora = 0.0f;
#pragma unroll
  for (int rr = 0; rr < RANK; ++rr)
    lora += lora_B[co * RANK + rr] * lora_A[rr * KDIM + ci * KW + k];
  acc += SCALING * lora;

  // Position in fragment-swizzled layout.
  const int kidx = k * C_IN + ci;          // global K index (k-major)
  const int c    = kidx >> 5;              // chunk
  const int kl   = kidx & 31;              // K within chunk
  int h, e;
  if (kl < 16) { h = kl >> 3;        e = kl & 7; }
  else         { h = (kl - 16) >> 3; e = 8 + ((kl - 16) & 7); }
  const int lane  = h * 16 + (co & 15);
  const int strip = co >> 4;
  W2f[(((size_t)strip * NCHUNK + c) * 32 + lane) * 16 + e] = f32_to_bf16(acc);
}

// ---------------------------------------------------------------------------
// Kernel 2: implicit-GEMM conv.
// Block: 256 threads = 8 waves. Tile: M=256 (all C_out), N=128 time cols.
// Each wave: M=32 (two 16-row strips), N=128 (eight 16-col tiles).
// K loop: 40 chunks of 32 (chunk c -> tap k=c>>3, ci window (c&7)*32).
// x slab staged once to LDS as bf16, t-major: slab[t][ci]; the whole K
// reduction reads only this slab (im2col = shift t by k, slice ci).
// ---------------------------------------------------------------------------
union Frag { v16bf v; u32x4 u[2]; };

__global__ __launch_bounds__(256) void conv_lora_wmma_kernel(
    const float* __restrict__ x,
    const unsigned short* __restrict__ W2f,
    const float* __restrict__ conv_b,
    float* __restrict__ out) {
  __shared__ alignas(16) unsigned short slab[SLAB_T * CPAD];

  const int tid  = threadIdx.x;
  const int wv   = tid >> 5;          // wave 0..7
  const int lane = tid & 31;
  const int l16  = lane & 15;
  const int h    = lane >> 4;         // lane half (WMMA layout selector)
  const int b    = blockIdx.y;        // batch
  const int t0   = blockIdx.x * NTILE;

  // ---- Stage x window [t0-2, t0+130) x all 256 ci into LDS as bf16 ----
  for (int ci = wv; ci < C_IN; ci += 8) {
    const float* xrow = x + ((size_t)(b * C_IN + ci)) * T_LEN;
    for (int t = lane; t < SLAB_T; t += 32) {
      const int gt = t0 - 2 + t;
      const float v = (gt >= 0 && gt < T_LEN) ? xrow[gt] : 0.0f;
      slab[t * CPAD + ci] = f32_to_bf16(v);
    }
  }
  __syncthreads();

  // ---- Per-lane A pointers into fragment-swizzled weights ----
  // Wave wv owns strips 2*wv and 2*wv+1. Per chunk: +512 ushorts (1 KB),
  // per-lane fragment is 16 contiguous ushorts (two b128 loads).
  const int m0 = wv * 32;
  const unsigned short* w2a = W2f + (size_t)(2 * wv) * STRIP_STRIDE + lane * 16;
  const unsigned short* w2b = w2a + STRIP_STRIDE;

  v8f acc[2][8];
#pragma unroll
  for (int mt = 0; mt < 2; ++mt)
#pragma unroll
    for (int nb = 0; nb < 8; ++nb) {
      v8f z = {};
      acc[mt][nb] = z;
    }

  for (int c = 0; c < NCHUNK; ++c) {
    const int k   = c >> 3;            // conv tap
    const int cio = (c & 7) * 32;      // ci window base

    Frag A0, A1;
    A0.u[0] = *(const u32x4*)(w2a + c * 512);
    A0.u[1] = *(const u32x4*)(w2a + c * 512 + 8);
    A1.u[0] = *(const u32x4*)(w2b + c * 512);
    A1.u[1] = *(const u32x4*)(w2b + c * 512 + 8);

    // B layout: elem e -> K_local = 8h + e (e<8) | 16+8h+(e-8) (e>=8) -> ci = cio + K_local; N = l16.
    const int brow0 = (l16 + k) * CPAD + cio + 8 * h;
#pragma unroll
    for (int nb = 0; nb < 8; ++nb) {
      Frag B;
      const unsigned short* bp = &slab[brow0 + nb * 16 * CPAD];
      B.u[0] = *(const u32x4*)(bp);
      B.u[1] = *(const u32x4*)(bp + 16);
      acc[0][nb] = __builtin_amdgcn_wmma_f32_16x16x32_bf16(
          false, A0.v, false, B.v, (short)0, acc[0][nb], false, false);
      acc[1][nb] = __builtin_amdgcn_wmma_f32_16x16x32_bf16(
          false, A1.v, false, B.v, (short)0, acc[1][nb], false, false);
    }
  }

  // ---- Store with bias (D layout: elem r -> M = 8h + r, N = l16) ----
#pragma unroll
  for (int mt = 0; mt < 2; ++mt) {
#pragma unroll
    for (int r = 0; r < 8; ++r) {
      const int co = m0 + mt * 16 + 8 * h + r;
      const float bias = conv_b[co];
      float* orow = out + ((size_t)(b * C_OUT + co)) * T_LEN + t0 + l16;
#pragma unroll
      for (int nb = 0; nb < 8; ++nb)
        orow[nb * 16] = acc[mt][nb][r] + bias;
    }
  }
}

// ---------------------------------------------------------------------------
extern "C" void kernel_launch(void* const* d_in, const int* in_sizes, int n_in,
                              void* d_out, int out_size, void* d_ws, size_t ws_size,
                              hipStream_t stream) {
  const float* x      = (const float*)d_in[0];
  const float* conv_w = (const float*)d_in[1];
  const float* conv_b = (const float*)d_in[2];
  const float* lora_A = (const float*)d_in[3];
  const float* lora_B = (const float*)d_in[4];
  float* out = (float*)d_out;
  unsigned short* W2f = (unsigned short*)d_ws;   // 256*1280 bf16 = 640 KB

  fuse_weights_kernel<<<dim3((C_OUT * KDIM) / 256), dim3(256), 0, stream>>>(
      conv_w, lora_A, lora_B, W2f);

  conv_lora_wmma_kernel<<<dim3(T_LEN / NTILE, NBATCH), dim3(256), 0, stream>>>(
      x, W2f, conv_b, out);
}
